// BoardModel_41197326303672
// MI455X (gfx1250) — compile-verified, matrix operand order checked by reading, not run on hardware
//
#include <hip/hip_runtime.h>
#include <hip/hip_bf16.h>

// ---------------------------------------------------------------------------
// Fused Tetris board-model kernel for MI455X (gfx1250, wave32, WMMA).
//
//  * Output-bandwidth bound (board output = 52 MB of 73 MB total traffic at
//    23.3 TB/s) -> single fused kernel: board built in LDS, streamed once
//    with float4 (b128) stores, LeNet+FC pipeline entirely in LDS.
//  * MHA degenerates (kv length 1 -> softmax == 1): a.mean(1) ==
//    out_proj(v_proj(b2)); emb / PE / q,k projections are dead and skipped.
//  * All GEMMs use v_wmma_f32_16x16x32_f16, M = 16-sample batch tile,
//    one 256-thread workgroup (8 wave32) per tile.
//  * Gathers are BRANCH-FREE: conv zero-padding is a halo in LDS; K-padding
//    selects load addresses into a zeroed LDS slot (v_cndmask, no exec
//    divergence).
//  * Fragment layout exploit: v16h elements 0..7 = K hi*8..hi*8+7 and
//    8..15 = K hi*8+16..+23 (two contiguous 16B groups) -> phases whose LDS
//    layout is unit-stride in K (conv3, lfc1, fc, v-proj, out-proj) load
//    whole A/B fragments as TWO ds_load_b128 each.  Strides are multiples
//    of 8 halves (16B alignment) chosen for 16 distinct banks per gather.
// ---------------------------------------------------------------------------

typedef __attribute__((ext_vector_type(16))) _Float16 v16h;
typedef __attribute__((ext_vector_type(8)))  _Float16 v8h;
typedef __attribute__((ext_vector_type(8)))  float    v8f;
typedef _Float16 f16;

#define WMMA_F16(a, b, c) \
    __builtin_amdgcn_wmma_f32_16x16x32_f16(false, (a), false, (b), (short)0, (c), false, false)

// 16-bit A-matrix 16x32 VGPR layout (CDNA5 ISA 7.12.2):
//   lanes 0-15:  v0..v3 hold K=0..7,  v4..v7 hold K=16..23
//   lanes 16-31: v0..v3 hold K=8..15, v4..v7 hold K=24..31
__device__ __forceinline__ int kslot(int v, int hi) {
    return (v < 4 ? 2 * v : 16 + 2 * (v - 4)) + (hi ? 8 : 0);
}

__device__ __forceinline__ v8h ld8(const f16* sm, int addr) {   // 16B-aligned
    return *(const v8h*)(sm + addr);
}
__device__ __forceinline__ v8h zero8() { v8h z = {}; return z; }
__device__ __forceinline__ v16h frag2(v8h lo, v8h hg) {
    v16h a;
    #pragma unroll
    for (int i = 0; i < 8; ++i) { a[i] = lo[i]; a[8 + i] = hg[i]; }
    return a;
}

// ---- LDS layout (in halves) ------------------------------------------------
#define SST1     1250                   // board sample stride (bank stride 49)
#define CST1     416                    // 26 * 16
#define SBP_OFF  0                      // 16 * 1250 = 20000
#define P1_OFF   20000                  // pooled conv1, stride 398 (6*66+2)
#define P1_SST   398
#define P2_OFF   (P1_OFF + 16 * P1_SST) // 26368 (16B aligned); stride 56
#define P2_SST   56
#define C3_OFF   (P2_OFF + 16 * P2_SST) // 27264; stride 72
#define C3_SST   72
#define B2H_OFF  (C3_OFF + 16 * C3_SST) // 28416; stride 40
#define B2_SST   40
#define VH_OFF   (B2H_OFF + 16 * B2_SST)// 29056; stride 40
#define TF_OFF   (VH_OFF + 16 * B2_SST) // 29696; t[:, :8] f16, stride 8
#define ZERO_OFF (TF_OFF + 128)         // 29824; 16 zeroed halves, 16B aligned
#define W1_OFF   (ZERO_OFF + 16)        // conv1_w 450 (pad 456)
#define B1_OFF   (W1_OFF + 456)         // conv1_b 6 (pad 8)
#define W2_OFF   (B1_OFF + 8)           // conv2_w 2400
#define BC2_OFF  (W2_OFF + 2400)        // conv2_b 16
#define W3_OFF   (BC2_OFF + 16)         // conv3_w 3072 (16B aligned)
#define B3_OFF   (W3_OFF + 3072)        // conv3_b 64
#define LW_OFF   (B3_OFF + 64)          // lfc1_w 2048 (16B aligned)
#define LB_OFF   (LW_OFF + 2048)        // lfc1_b 32
#define FW_OFF   (LB_OFF + 32)          // fc_w 256 (16B aligned)
#define FB_OFF   (FW_OFF + 256)         // fc_b 32
#define VW_OFF   (FB_OFF + 32)          // v-proj w 1024 (16B aligned)
#define VB_OFF   (VW_OFF + 1024)        // v-proj b 32
#define OW_OFF   (VB_OFF + 32)          // out_proj_w 1024 (16B aligned)
#define OB_OFF   (OW_OFF + 1024)        // out_proj_b 32
#define SMEM_HALVES (OB_OFF + 32)

__device__ __forceinline__ void load_w(f16* dst, const float* src, int n,
                                       int tid, int nthr) {
    for (int i = tid; i < n; i += nthr) dst[i] = (f16)src[i];
}

__global__ __launch_bounds__(256, 2)
void board_model_fused(const int* __restrict__ t,
                       const int* __restrict__ piece_table,
                       const float* __restrict__ c1w, const float* __restrict__ c1b,
                       const float* __restrict__ c2w, const float* __restrict__ c2b,
                       const float* __restrict__ c3w, const float* __restrict__ c3b,
                       const float* __restrict__ lw,  const float* __restrict__ lb,
                       const float* __restrict__ fw,  const float* __restrict__ fb,
                       const float* __restrict__ inw, const float* __restrict__ inb,
                       const float* __restrict__ ow,  const float* __restrict__ ob,
                       float* __restrict__ out, int B) {
    extern __shared__ f16 sm[];

    const int tid  = threadIdx.x;
    const int lane = tid & 31;
    const int wave = tid >> 5;
    const int hi   = lane >> 4;
    const int n    = lane & 15;     // N column within a 16-wide tile
    const int sl   = lane & 15;     // sample row within the M tile
    const int s0   = blockIdx.x * 16;

    float* out_mid   = out;                   // (B, 96)
    float* out_board = out + (size_t)B * 96;  // (B, 3*22*12)

    // Warm L2 with this block's rows of t (emits global_prefetch_b8).
    if (lane == 0) __builtin_prefetch(t + (size_t)(s0 + wave * 2) * 232, 0, 3);

    // ---- Phase 0a: weights -> LDS (f16), zero slot, t[:, :8] staging -------
    load_w(sm + W1_OFF,  c1w, 450,  tid, 256);
    load_w(sm + B1_OFF,  c1b, 6,    tid, 256);
    load_w(sm + W2_OFF,  c2w, 2400, tid, 256);
    load_w(sm + BC2_OFF, c2b, 16,   tid, 256);
    load_w(sm + W3_OFF,  c3w, 3072, tid, 256);
    load_w(sm + B3_OFF,  c3b, 64,   tid, 256);
    load_w(sm + LW_OFF,  lw,  2048, tid, 256);
    load_w(sm + LB_OFF,  lb,  32,   tid, 256);
    load_w(sm + FW_OFF,  fw,  256,  tid, 256);
    load_w(sm + FB_OFF,  fb,  32,   tid, 256);
    load_w(sm + VW_OFF,  inw + 64 * 32, 1024, tid, 256);  // v-projection rows
    load_w(sm + VB_OFF,  inb + 64,      32,   tid, 256);
    load_w(sm + OW_OFF,  ow,  1024, tid, 256);
    load_w(sm + OB_OFF,  ob,  32,   tid, 256);
    if (tid < 16) sm[ZERO_OFF + tid] = (f16)0.0f;
    if (tid < 128) {                               // t[:, :8] as f16
        int s = tid >> 3, j = tid & 7;
        sm[TF_OFF + s * 8 + j] = (f16)(float)t[(s0 + s) * 232 + j];
    }

    // ---- Phase 0b: board with conv1 halo (16 x 3 x 26 x 16) ----------------
    for (int i = tid; i < 16 * 3 * 26 * 16; i += 256) {
        int col = i & 15;
        int row = (i >> 4) % 26;
        int ch  = (i / (26 * 16)) % 3;
        int s   = i / (26 * 16 * 3);
        int iy = row - 2, ix = col - 2;
        f16 v = (f16)0.0f;
        if (iy >= 0 && iy < 22 && ix >= 0 && ix < 12) {
            if (iy == 21 || ix == 0 || ix == 11) v = (f16)1.0f;
            else if (ch == 0)
                v = (f16)(float)t[(s0 + s) * 232 + 22 + iy * 10 + (ix - 1)];
        }
        sm[SBP_OFF + s * SST1 + ch * CST1 + row * 16 + col] = v;
    }
    __syncthreads();

    // ---- Phase 0c: piece placement (one thread per sample) -----------------
    if (tid < 16) {
        const int* tr = t + (s0 + tid) * 232;
        const int* pt = piece_table + (tr[8] * 4 + tr[4]) * 16;
        int idx4[4]; int c = 0;
        for (int i = 0; i < 16; ++i) if (pt[i])  { if (c < 4) idx4[c++] = i; }
        for (int i = 0; i < 16 && c < 4; ++i) if (!pt[i]) idx4[c++] = i;
        for (int i = 0; i < 4; ++i) {
            int cy = idx4[i] >> 2, cx = idx4[i] & 3;
            int x  = cx + tr[1] - 2;       // unpadded board col
            int y  = cy + tr[2];
            int ny = y + tr[3];
            bool m = (y >= 0) && (ny >= 0);
            if (m && y  < 21 && x >= 0 && x < 10)
                sm[SBP_OFF + tid * SST1 + 1 * CST1 + (y  + 2) * 16 + (x + 3)] = (f16)1.0f;
            if (m && ny < 21 && x >= 0 && x < 10)
                sm[SBP_OFF + tid * SST1 + 2 * CST1 + (ny + 2) * 16 + (x + 3)] = (f16)1.0f;
        }
    }
    __syncthreads();

    // ---- Phase 0d: stream board to HBM via float4 (b128) stores ------------
    for (int i = tid; i < 16 * 198; i += 256) {     // 792/4 = 198 quads
        int s  = i / 198, q = i % 198;
        int c  = q * 4;
        int ch = c / 264, rem = c % 264;
        int y  = rem / 12, x = rem % 12;            // 12 % 4 == 0: same row
        int a  = SBP_OFF + s * SST1 + ch * CST1 + (y + 2) * 16 + (x + 2);
        float4 v = make_float4((float)sm[a], (float)sm[a + 1],
                               (float)sm[a + 2], (float)sm[a + 3]);
        *(float4*)(out_board + (size_t)(s0 + s) * 792 + c) = v;
    }

    // ---- Phase 1: conv1 (5x5 pad2, 3->6) + ReLU + avgpool, WMMA ------------
    // per pixel: M=16 samples, K=75->96 (3 k-steps), N=6-in-16
    {
        v16h bf[3];
        #pragma unroll
        for (int ks = 0; ks < 3; ++ks) {
            v16h b;
            #pragma unroll
            for (int v = 0; v < 8; ++v) {
                #pragma unroll
                for (int e = 0; e < 2; ++e) {
                    int k = ks * 32 + kslot(v, hi) + e;
                    int ic = k / 25, rem = k % 25, r = rem / 5, cc = rem % 5;
                    bool valid = (k < 75) && (n < 6);
                    int wi = valid ? ((n * 3 + ic) * 5 + r) * 5 + cc : 0;
                    f16 w = sm[W1_OFF + wi];
                    b[2 * v + e] = valid ? w : (f16)0.0f;
                }
            }
            bf[ks] = b;
        }
        float bias = (n < 6) ? (float)sm[B1_OFF + n] : 0.0f;
        for (int win = wave; win < 66; win += 8) {
            int py = win / 6, px = win % 6;
            v8f acc = {};
            #pragma unroll
            for (int d = 0; d < 4; ++d) {
                int oy = 2 * py + (d >> 1), ox = 2 * px + (d & 1);
                int baseA = SBP_OFF + sl * SST1 + oy * 16 + ox; // row=oy+r col=ox+cc
                v8f cfrag = {};
                #pragma unroll
                for (int ks = 0; ks < 3; ++ks) {
                    v16h a;
                    #pragma unroll
                    for (int v = 0; v < 8; ++v) {
                        #pragma unroll
                        for (int e = 0; e < 2; ++e) {
                            int k = ks * 32 + kslot(v, hi) + e;
                            int ic = k / 25, rem = k % 25, r = rem / 5, cc = rem % 5;
                            int addr = baseA + ic * CST1 + r * 16 + cc;
                            addr = (k < 75) ? addr : ZERO_OFF;   // cndmask
                            a[2 * v + e] = sm[addr];
                        }
                    }
                    cfrag = WMMA_F16(a, bf[ks], cfrag);
                }
                #pragma unroll
                for (int v = 0; v < 8; ++v) {
                    float r = cfrag[v] + bias;
                    acc[v] += r > 0.0f ? r : 0.0f;
                }
            }
            #pragma unroll
            for (int v = 0; v < 8; ++v) {
                int m = v + hi * 8;
                if (n < 6) sm[P1_OFF + m * P1_SST + n * 66 + win] = (f16)(acc[v] * 0.25f);
            }
        }
    }
    __syncthreads();

    // ---- Phase 2: conv2 (5x5 valid, 6->16) + ReLU + avgpool ----------------
    if (wave < 3) {
        v16h bf[5];
        #pragma unroll
        for (int ks = 0; ks < 5; ++ks) {
            v16h b;
            #pragma unroll
            for (int v = 0; v < 8; ++v) {
                #pragma unroll
                for (int e = 0; e < 2; ++e) {
                    int k = ks * 32 + kslot(v, hi) + e;
                    int ic = k / 25, rem = k % 25, r = rem / 5, cc = rem % 5;
                    bool valid = (k < 150);
                    int wi = valid ? ((n * 6 + ic) * 5 + r) * 5 + cc : 0;
                    f16 w = sm[W2_OFF + wi];
                    b[2 * v + e] = valid ? w : (f16)0.0f;
                }
            }
            bf[ks] = b;
        }
        float bias = (float)sm[BC2_OFF + n];
        int win = wave;                 // pooled row 0..2 of (7,2)->(3,1)
        v8f acc = {};
        #pragma unroll
        for (int d = 0; d < 4; ++d) {
            int oy = 2 * win + (d >> 1), ox = d & 1;
            int baseA = P1_OFF + sl * P1_SST + oy * 6 + ox;
            v8f cfrag = {};
            #pragma unroll
            for (int ks = 0; ks < 5; ++ks) {
                v16h a;
                #pragma unroll
                for (int v = 0; v < 8; ++v) {
                    #pragma unroll
                    for (int e = 0; e < 2; ++e) {
                        int k = ks * 32 + kslot(v, hi) + e;
                        int ic = k / 25, rem = k % 25, r = rem / 5, cc = rem % 5;
                        int addr = baseA + ic * 66 + r * 6 + cc;
                        addr = (k < 150) ? addr : ZERO_OFF;
                        a[2 * v + e] = sm[addr];
                    }
                }
                cfrag = WMMA_F16(a, bf[ks], cfrag);
            }
            #pragma unroll
            for (int v = 0; v < 8; ++v) {
                float r = cfrag[v] + bias;
                acc[v] += r > 0.0f ? r : 0.0f;
            }
        }
        #pragma unroll
        for (int v = 0; v < 8; ++v) {
            int m = v + hi * 8;
            sm[P2_OFF + m * P2_SST + n * 3 + win] = (f16)(acc[v] * 0.25f);
        }
    }
    __syncthreads();

    // ---- Phase 3: conv3 (3x1 valid, 16->64) + ReLU -------------------------
    // K = ic*3+ky (unit stride in P2 and W3): A and B fragments are b128
    // group loads.  k-step 1 upper group (K 48..63) is compile-time zero.
    if (wave < 4) {
        int oc = wave * 16 + n;
        int wb = W3_OFF + oc * 48;
        int pa = P2_OFF + sl * P2_SST;
        v16h b0 = frag2(ld8(sm, wb + hi * 8),      ld8(sm, wb + hi * 8 + 16));
        v16h b1 = frag2(ld8(sm, wb + 32 + hi * 8), zero8());
        v16h a0 = frag2(ld8(sm, pa + hi * 8),      ld8(sm, pa + hi * 8 + 16));
        v16h a1 = frag2(ld8(sm, pa + 32 + hi * 8), zero8());
        v8f cfrag = {};
        cfrag = WMMA_F16(a0, b0, cfrag);
        cfrag = WMMA_F16(a1, b1, cfrag);
        float bias = (float)sm[B3_OFF + oc];
        #pragma unroll
        for (int v = 0; v < 8; ++v) {
            int m = v + hi * 8;
            float r = cfrag[v] + bias;
            sm[C3_OFF + m * C3_SST + oc] = (f16)(r > 0.0f ? r : 0.0f);
        }
    }
    __syncthreads();

    // ---- Phase 4: lfc1 (64->32, ReLU) => b2 ; fc (8->32, ReLU) => xf -------
    if (wave < 2) {              // b2 = relu(c3 @ lfc1_w.T + lfc1_b)
        int ng = wave * 16 + n;
        int ca = C3_OFF + sl * C3_SST;
        int la = LW_OFF + ng * 64;
        v8f cfrag = {};
        #pragma unroll
        for (int ks = 0; ks < 2; ++ks) {
            int ka = ks * 32 + hi * 8;
            v16h a = frag2(ld8(sm, ca + ka), ld8(sm, ca + ka + 16));
            v16h b = frag2(ld8(sm, la + ka), ld8(sm, la + ka + 16));
            cfrag = WMMA_F16(a, b, cfrag);
        }
        float bias = (float)sm[LB_OFF + ng];
        #pragma unroll
        for (int v = 0; v < 8; ++v) {
            int m = v + hi * 8;
            float r = cfrag[v] + bias;
            r = r > 0.0f ? r : 0.0f;
            sm[B2H_OFF + m * B2_SST + ng] = (f16)r;
            out_mid[(size_t)(s0 + m) * 96 + 64 + ng] = r;   // cols 64..95: b2
        }
    } else if (wave < 4) {       // xf = relu(t[:, :8] @ fc_w.T + fc_b)
        int ng = (wave - 2) * 16 + n;
        // valid K group only for lanes 0-15 (K 0..7); others -> zero slot
        int a0 = hi ? ZERO_OFF : (TF_OFF + sl * 8);
        int b0 = hi ? ZERO_OFF : (FW_OFF + ng * 8);
        v16h a = frag2(ld8(sm, a0), zero8());
        v16h b = frag2(ld8(sm, b0), zero8());
        v8f cfrag = {};
        cfrag = WMMA_F16(a, b, cfrag);
        float bias = (float)sm[FB_OFF + ng];
        #pragma unroll
        for (int v = 0; v < 8; ++v) {
            int m = v + hi * 8;
            float r = cfrag[v] + bias;
            out_mid[(size_t)(s0 + m) * 96 + ng] = r > 0.0f ? r : 0.0f; // cols 0..31
        }
    }
    __syncthreads();

    // ---- Phase 5: collapsed MHA: a = out_proj(v_proj(b2)) ------------------
    if (wave < 2) {              // vh = b2 @ vw.T + vb
        int ng = wave * 16 + n;
        int ba = B2H_OFF + sl * B2_SST + hi * 8;
        int wb = VW_OFF + ng * 32 + hi * 8;
        v16h a = frag2(ld8(sm, ba), ld8(sm, ba + 16));
        v16h b = frag2(ld8(sm, wb), ld8(sm, wb + 16));
        v8f cfrag = {};
        cfrag = WMMA_F16(a, b, cfrag);
        float bias = (float)sm[VB_OFF + ng];
        #pragma unroll
        for (int v = 0; v < 8; ++v) {
            int m = v + hi * 8;
            sm[VH_OFF + m * B2_SST + ng] = (f16)(cfrag[v] + bias);
        }
    }
    __syncthreads();
    if (wave < 2) {              // a = vh @ out_proj_w.T + out_proj_b
        int ng = wave * 16 + n;
        int va = VH_OFF + sl * B2_SST + hi * 8;
        int wb = OW_OFF + ng * 32 + hi * 8;
        v16h a = frag2(ld8(sm, va), ld8(sm, va + 16));
        v16h b = frag2(ld8(sm, wb), ld8(sm, wb + 16));
        v8f cfrag = {};
        cfrag = WMMA_F16(a, b, cfrag);
        float bias = (float)sm[OB_OFF + ng];
        #pragma unroll
        for (int v = 0; v < 8; ++v) {
            int m = v + hi * 8;
            out_mid[(size_t)(s0 + m) * 96 + 32 + ng] = cfrag[v] + bias; // cols 32..63
        }
    }
}

extern "C" void kernel_launch(void* const* d_in, const int* in_sizes, int n_in,
                              void* d_out, int out_size, void* d_ws, size_t ws_size,
                              hipStream_t stream) {
    (void)in_sizes; (void)n_in; (void)d_ws; (void)ws_size; (void)out_size;
    const int B = 16384;
    const int*   t   = (const int*)  d_in[0];
    const int*   pt  = (const int*)  d_in[1];
    const float* c1w = (const float*)d_in[2];
    const float* c1b = (const float*)d_in[3];
    const float* c2w = (const float*)d_in[4];
    const float* c2b = (const float*)d_in[5];
    const float* c3w = (const float*)d_in[6];
    const float* c3b = (const float*)d_in[7];
    const float* lw  = (const float*)d_in[8];
    const float* lb  = (const float*)d_in[9];
    const float* fw  = (const float*)d_in[10];
    const float* fb  = (const float*)d_in[11];
    // d_in[12] = emb : dead (softmax over a single key is identically 1.0)
    const float* inw = (const float*)d_in[13];
    const float* inb = (const float*)d_in[14];
    const float* ow  = (const float*)d_in[15];
    const float* ob  = (const float*)d_in[16];

    dim3 grid(B / 16), block(256);
    size_t smem = (size_t)SMEM_HALVES * sizeof(_Float16);
    board_model_fused<<<grid, block, smem, stream>>>(
        t, pt, c1w, c1b, c2w, c2b, c3w, c3b, lw, lb, fw, fb,
        inw, inb, ow, ob, (float*)d_out, B);
}
